// QuantMatMul_31971736552073
// MI455X (gfx1250) — 8-bit matrix operand order repaired and checked by reading, compiled, not run on hardware
//
#include <hip/hip_runtime.h>
#include <stdint.h>


#define BH 128
#define Sdim 1024
#define Ddim 64
#define PLANE 65536

typedef int   v8i  __attribute__((ext_vector_type(8)));
typedef int   v4i  __attribute__((ext_vector_type(4)));
typedef int   v2i  __attribute__((ext_vector_type(2)));
typedef float v8f  __attribute__((ext_vector_type(8)));

__device__ __forceinline__ unsigned umaxu(unsigned a, unsigned b) { return a > b ? a : b; }

__global__ void init_ws_kernel(unsigned* __restrict__ scl) {
    if (blockIdx.x == 0 && threadIdx.x == 0) { scl[0] = 0u; scl[1] = 0u; }
}

__global__ void absmax_kernel(const float* __restrict__ x, long n4,
                              unsigned* __restrict__ out) {
    __shared__ unsigned red[256];
    unsigned m = 0u;
    const float4* x4 = (const float4*)x;
    for (long i = (long)blockIdx.x * blockDim.x + threadIdx.x; i < n4;
         i += (long)gridDim.x * blockDim.x) {
        float4 v = x4[i];
        m = umaxu(m, __float_as_uint(fabsf(v.x)));
        m = umaxu(m, __float_as_uint(fabsf(v.y)));
        m = umaxu(m, __float_as_uint(fabsf(v.z)));
        m = umaxu(m, __float_as_uint(fabsf(v.w)));
    }
    red[threadIdx.x] = m;
    __syncthreads();
    for (int s = 128; s > 0; s >>= 1) {
        if ((int)threadIdx.x < s) red[threadIdx.x] = umaxu(red[threadIdx.x], red[threadIdx.x + s]);
        __syncthreads();
    }
    if (threadIdx.x == 0) atomicMax(out, red[0]);
}

__device__ __forceinline__ unsigned q_to_fp8(float x, float inv) {
    int q = (int)rintf(x * inv);
    q = q < -2 ? -2 : (q > 1 ? 1 : q);
    unsigned e = (q == 1) ? 0x38u : 0x00u;
    e = (q == -1) ? 0xB8u : e;
    e = (q == -2) ? 0xC0u : e;
    return e;
}

__device__ __forceinline__ int quant_pack4(float4 v, float inv) {
    return (int)(q_to_fp8(v.x, inv)       | (q_to_fp8(v.y, inv) << 8) |
                 (q_to_fp8(v.z, inv) << 16) | (q_to_fp8(v.w, inv) << 24));
}

__global__ void quantA_kernel(const float* __restrict__ x, unsigned char* __restrict__ q,
                              const unsigned* __restrict__ scl, long n4) {
    float s = __uint_as_float(scl[0]);
    float inv = s > 0.f ? 1.f / s : 0.f;
    const float4* x4 = (const float4*)x;
    int* q4 = (int*)q;
    for (long i = (long)blockIdx.x * blockDim.x + threadIdx.x; i < n4;
         i += (long)gridDim.x * blockDim.x)
        q4[i] = quant_pack4(x4[i], inv);
}

__global__ void quantB_kernel(const float* __restrict__ x, unsigned char* __restrict__ qt,
                              const unsigned* __restrict__ scl) {
    float s = __uint_as_float(scl[1]);
    float inv = s > 0.f ? 1.f / s : 0.f;
    long tid = (long)blockIdx.x * blockDim.x + threadIdx.x;
    int b  = (int)(tid >> 14);
    int r  = (int)(tid & 16383);
    int k4 = r >> 10;
    int n  = r & 1023;
    const float* base = x + ((long)b << 16) + ((long)(k4 * 4) << 10) + n;
    float4 v;
    v.x = base[0];
    v.y = base[1024];
    v.z = base[2048];
    v.w = base[3072];
    *(int*)(qt + ((long)b << 16) + (long)n * 64 + k4 * 4) = quant_pack4(v, inv);
}

__global__ void __launch_bounds__(256) gemm_kernel(const unsigned char* __restrict__ Aq,
                                                   const unsigned char* __restrict__ Bt,
                                                   const unsigned* __restrict__ scl,
                                                   float* __restrict__ out) {
    int wid  = blockIdx.x * 8 + (threadIdx.x >> 5);
    int lane = threadIdx.x & 31;
    int b    = wid >> 8;
    int rem  = wid & 255;
    int m    = rem >> 2;
    int c    = rem & 3;
    int g    = lane >> 4;
    int lm   = lane & 15;

    float scale = __uint_as_float(scl[0]) * __uint_as_float(scl[1]);

    const unsigned char* arow = Aq + ((long)b << 16) + (long)(m * 16 + lm) * 64 + g * 16;
    v2i p0 = *(const v2i*)(arow +  0);
    v2i p1 = *(const v2i*)(arow +  8);
    v2i p2 = *(const v2i*)(arow + 32);
    v2i p3 = *(const v2i*)(arow + 40);
    v8i a = {p0.x, p0.y, p1.x, p1.y, p2.x, p2.y, p3.x, p3.y};

    const unsigned char* bbase = Bt + ((long)b << 16);
    int row0 = m * 16 + g * 8;
    float* obase = out + ((long)b << 20) + (long)row0 * 1024 + lm;

    const v8f czero = {0.f, 0.f, 0.f, 0.f, 0.f, 0.f, 0.f, 0.f};

    for (int t = 0; t < 16; ++t) {
        int n0 = (c << 8) + (t << 4);
        const unsigned char* brow = bbase + (long)(n0 + lm) * 64 + g * 16;
        v4i lo = *(const v4i*)(brow);
        v4i hi = *(const v4i*)(brow + 32);
        v8i bm = {lo.x, lo.y, lo.z, lo.w, hi.x, hi.y, hi.z, hi.w};
        if (t < 15) __builtin_prefetch(brow + 16 * 64, 0, 1);

        v8f acc = __builtin_amdgcn_wmma_f32_16x16x64_fp8_fp8(a, bm, (short)0,
                                                             czero, false, false);

        float* op = obase + n0;
        #pragma unroll
        for (int r = 0; r < 8; ++r)
            __builtin_nontemporal_store(acc[r] * scale, op + (long)r * 1024);
    }
}

extern "C" void kernel_launch(void* const* d_in, const int* in_sizes, int n_in,
                              void* d_out, int out_size, void* d_ws, size_t ws_size,
                              hipStream_t stream) {
    const float* A = (const float*)d_in[0];
    const float* B = (const float*)d_in[1];
    float* out = (float*)d_out;

    unsigned char* ws = (unsigned char*)d_ws;
    unsigned*      scl = (unsigned*)ws;
    unsigned char* Aq  = ws + 256;
    unsigned char* Bt  = ws + 256 + (size_t)BH * PLANE;

    long nA4 = (long)in_sizes[0] / 4;
    long nB4 = (long)in_sizes[1] / 4;

    hipLaunchKernelGGL(init_ws_kernel, dim3(1), dim3(64), 0, stream, scl);
    hipLaunchKernelGGL(absmax_kernel, dim3(1024), dim3(256), 0, stream, A, nA4, scl + 0);
    hipLaunchKernelGGL(absmax_kernel, dim3(1024), dim3(256), 0, stream, B, nB4, scl + 1);
    hipLaunchKernelGGL(quantA_kernel, dim3(2048), dim3(256), 0, stream, A, Aq, scl, nA4);
    hipLaunchKernelGGL(quantB_kernel, dim3(8192), dim3(256), 0, stream, B, Bt, scl);
    hipLaunchKernelGGL(gemm_kernel, dim3(4096), dim3(256), 0, stream, Aq, Bt, scl, out);
}
